// PFA_75505525064035
// MI455X (gfx1250) — matrix operand order repaired and checked by reading, not compiled, not run on hardware
//
#include <hip/hip_runtime.h>
#include <math.h>

// CDNA5 / gfx1250: wave32, WMMA bf16 16x16x32 path.
typedef __attribute__((ext_vector_type(16))) __bf16 v16bf;
typedef __attribute__((ext_vector_type(8)))  float  v8f;
typedef __attribute__((ext_vector_type(8)))  int    v8i;

#define S_T  20
#define N_T  102400
#define B_T  1024
#define EMB  64
#define MAXR 128   // up to 8 row-tiles of 16 -> 8 waves

// Branch-free tanh from native transcendentals:
//   tanh(x) = 1 - 2/(exp2(2*log2e*x) + 1)
// v_exp_f32 overflow -> inf -> rcp -> 0 -> +1 ; underflow -> 0 -> rcp(1) -> -1.
__device__ __forceinline__ float fast_tanh(float x) {
    const float e = __builtin_amdgcn_exp2f(x * 2.8853900817779268f); // 2*log2(e)
    const float r = __builtin_amdgcn_rcpf(e + 1.0f);                 // v_rcp_f32
    return fmaf(-2.0f, r, 1.0f);
}

// Dynamic LDS layout (bytes):
//  hbf   : MAXR*EMB bf16     (bf16 shadow of this step's h -> WMMA A for h@W_out)
//  gmbf  : MAXR*EMB bf16     (bf16 shadow of gm_sum -> WMMA A for gm@W_gm; f32
//                             master copy of gm_sum lives in registers, C layout)
//  wgm   : EMB*EMB bf16      (W_gm staged once, k-major)
//  win0/win1/bhs : EMB f32 each
//  maskf/cxs/cys : MAXR f32 each
//  red   : 4 f32 (cnt, ssx, ssy, pad)
//  ired  : 256 i32 (pednum prefix-sum reduce)
constexpr size_t SMEM_BYTES =
    (size_t)MAXR * EMB * 2 * 2 +   // hbf + gmbf (bf16)
    (size_t)EMB * EMB * 2 +        // wgm bf16
    (size_t)(EMB * 3 + MAXR * 3 + 4) * 4 +
    256 * 4;

__global__ __launch_bounds__(256) void pfa_scan_kernel(
    const float* __restrict__ nodes_abs,   // (S,N,2)
    const float* __restrict__ seq_list,    // (S,N)
    const int*   __restrict__ pednum,      // (B)
    const float* __restrict__ W_in,        // (2,EMB)
    const float* __restrict__ W_gm,        // (EMB,EMB) row-major: [k][n]
    const float* __restrict__ b_h,         // (EMB)
    const float* __restrict__ W_out,       // (EMB,2)
    const float* __restrict__ b_out,       // (2)
    float* __restrict__ out)               // (S,N,2)
{
    extern __shared__ unsigned char smem[];
    __bf16* hbf   = (__bf16*)smem;
    __bf16* gmbf  = hbf + MAXR * EMB;
    __bf16* wgm   = gmbf + MAXR * EMB;
    float*  win0  = (float*)(wgm + EMB * EMB);
    float*  win1  = win0 + EMB;
    float*  bhs   = win1 + EMB;
    float*  maskf = bhs + EMB;
    float*  cxs   = maskf + MAXR;
    float*  cys   = cxs + MAXR;
    float*  red   = cys + MAXR;
    int*    ired  = (int*)(red + 4);

    const int tid  = threadIdx.x;
    const int lane = tid & 31;
    const int wave = tid >> 5;
    const int bid  = blockIdx.x;
    const int half = lane >> 4;      // which half-wave (K/M offset selector)
    const int lrow = lane & 15;

    // ---- prefix sum of pednum -> this scene's row range ----
    int part = 0;
    for (int i = tid; i < bid; i += 256) part += pednum[i];
    ired[tid] = part;
    __syncthreads();
    for (int s = 128; s > 0; s >>= 1) {
        if (tid < s) ired[tid] += ired[tid + s];
        __syncthreads();
    }
    const int start = ired[0];
    int rows = pednum[bid];
    if (rows > MAXR) rows = MAXR;
    const int ntiles = (rows + 15) >> 4;
    const bool active = (wave < ntiles);

    // ---- stage weights / init state ----
    for (int i = tid; i < EMB * EMB; i += 256) wgm[i] = (__bf16)W_gm[i];
    for (int i = tid; i < (MAXR * EMB) / 2; i += 256) ((unsigned int*)gmbf)[i] = 0u;
    for (int i = tid; i < EMB; i += 256) {
        win0[i] = W_in[i];
        win1[i] = W_in[EMB + i];
        bhs[i]  = b_h[i];
    }
    for (int i = tid; i < MAXR; i += 256) {
        maskf[i] = (i < rows) ? 1.0f : 0.0f;
        cxs[i] = 0.0f;
        cys[i] = 0.0f;
    }
    __syncthreads();

    // ---- W_gm B-fragments, loaded once per wave ----
    // B 32x16 bf16: lane = column n (mod 16); VGPR j holds K = 2j + 16*half (+ kf*32).
    v16bf Bf[4][2];
#pragma unroll
    for (int t = 0; t < 4; ++t) {
        const int n = t * 16 + lrow;
#pragma unroll
        for (int kf = 0; kf < 2; ++kf) {
#pragma unroll
            for (int j = 0; j < 8; ++j) {
                const int k = 2 * j + half * 16 + kf * 32;
                Bf[t][kf][2 * j + 0] = wgm[(k + 0) * EMB + n];
                Bf[t][kf][2 * j + 1] = wgm[(k + 1) * EMB + n];
            }
        }
    }

    // ---- W_out padded to 64x16 (cols 2..15 zero) as B-fragments, built once ----
    v16bf Bo[2];
    {
        const bool live = (lrow < 2);
#pragma unroll
        for (int kf = 0; kf < 2; ++kf) {
#pragma unroll
            for (int j = 0; j < 8; ++j) {
                const int k = 2 * j + half * 16 + kf * 32;
                Bo[kf][2 * j + 0] = (__bf16)(live ? W_out[(k + 0) * 2 + lrow] : 0.0f);
                Bo[kf][2 * j + 1] = (__bf16)(live ? W_out[(k + 1) * 2 + lrow] : 0.0f);
            }
        }
    }
    const float boutv = (lrow < 2) ? b_out[lrow] : 0.0f;

    const int tb = wave << 4;   // row-tile base
    const int M  = tb + lrow;   // this lane's A-operand row (< MAXR)

    // gm_sum master copy: f32, C-fragment layout (4 column tiles x 8 rows/lane).
    v8f gmreg[4] = {};

    // ================= sequential scan over time =================
    for (int f = 0; f < S_T - 1; ++f) {
        if (tid == 0) { red[0] = 0.0f; red[1] = 0.0f; red[2] = 0.0f; }
        __syncthreads();

        // -- mask update + scene reduction (ds_add_f32 atomics) --
        float ax = 0.0f, ay = 0.0f, mf = 0.0f;
        if (tid < rows) {
            const int g = start + tid;
            const float sq = seq_list[(size_t)f * N_T + g];
            mf = maskf[tid] * (sq > 0.0f ? 1.0f : 0.0f);
            maskf[tid] = mf;
            ax = nodes_abs[(size_t)f * N_T * 2 + (size_t)g * 2 + 0];
            ay = nodes_abs[(size_t)f * N_T * 2 + (size_t)g * 2 + 1];
            atomicAdd(&red[0], mf);
            atomicAdd(&red[1], ax * mf);
            atomicAdd(&red[2], ay * mf);
        }
        __syncthreads();

        // -- centered, masked coordinates --
        if (tid < rows) {
            const float cnt = red[0];
            const float dinv = __builtin_amdgcn_rcpf(cnt > 1.0f ? cnt : 1.0f);
            cxs[tid] = (ax - red[1] * dinv) * mf;
            cys[tid] = (ay - red[2] * dinv) * mf;
        }
        __syncthreads();

        const float invf = __builtin_amdgcn_rcpf((float)(f > 1 ? f : 1));

        if (active) {
            // -- A-fragments: raw b32 loads of the bf16 gm_sum shadow (two bf16
            //    per dword == one A-fragment VGPR). Intra-wave RAW on LDS only.
            // A 16x32 bf16: lane row = M; VGPR j -> K pair 2j + 16*half.
            v16bf A0, A1;
            {
                const unsigned int* gw = (const unsigned int*)(gmbf + M * EMB);
                v8i a0i, a1i;
#pragma unroll
                for (int j = 0; j < 8; ++j) {
                    const int w = j + half * 8;  // dword index
                    a0i[j] = gw[w];
                    a1i[j] = gw[w + 16];
                }
                A0 = __builtin_bit_cast(v16bf, a0i);
                A1 = __builtin_bit_cast(v16bf, a1i);
            }

            // Hoist per-row operands (m = tb + j + 8*half, independent of t).
            float lcx[8], lcy[8], lmf[8];
#pragma unroll
            for (int j = 0; j < 8; ++j) {
                const int m = tb + j + half * 8;
                lcx[j] = cxs[m];
                lcy[j] = cys[m];
                lmf[j] = maskf[m];
            }

            // -- WMMA: (16 x 64) @ (64 x 64) per wave, K split in two 32-chunks --
#pragma unroll
            for (int t = 0; t < 4; ++t) {
                v8f acc = {};
                acc = __builtin_amdgcn_wmma_f32_16x16x32_bf16(
                          false, A0, false, Bf[t][0], (short)0, acc, false, false);
                acc = __builtin_amdgcn_wmma_f32_16x16x32_bf16(
                          false, A1, false, Bf[t][1], (short)0, acc, false, false);
                // C 16x16 f32: lane = column (mod 16); VGPR j -> M = j + 8*half.
                const int n = t * 16 + lrow;
                const float w0 = win0[n], w1 = win1[n], bb = bhs[n];
#pragma unroll
                for (int j = 0; j < 8; ++j) {
                    const int m = tb + j + half * 8;
                    const float lin = fmaf(lcx[j], w0, fmaf(lcy[j], w1, bb));
                    const float v = fmaf(acc[j], invf, lin);
                    const float hv = fast_tanh(v) * lmf[j];
                    gmreg[t][j] += hv;                       // f32 recurrent state
                    gmbf[m * EMB + n] = (__bf16)gmreg[t][j]; // bf16 shadow (b16 store)
                    hbf[m * EMB + n]  = (__bf16)hv;          // bf16 h shadow
                }
            }

            // -- output projection on the matrix unit: (16x64) @ (64x16-padded) --
            // hbf rows of this wave were written by this wave: intra-wave in-order LDS.
            {
                const unsigned int* hw = (const unsigned int*)(hbf + M * EMB);
                v8i h0i, h1i;
#pragma unroll
                for (int j = 0; j < 8; ++j) {
                    const int w = j + half * 8;
                    h0i[j] = hw[w];
                    h1i[j] = hw[w + 16];
                }
                const v16bf H0 = __builtin_bit_cast(v16bf, h0i);
                const v16bf H1 = __builtin_bit_cast(v16bf, h1i);
                v8f po = {};
                po = __builtin_amdgcn_wmma_f32_16x16x32_bf16(
                         false, H0, false, Bo[0], (short)0, po, false, false);
                po = __builtin_amdgcn_wmma_f32_16x16x32_bf16(
                         false, H1, false, Bo[1], (short)0, po, false, false);
                // Only columns 0,1 are meaningful: lanes {0,1,16,17} store.
                if (lrow < 2) {
#pragma unroll
                    for (int j = 0; j < 8; ++j) {
                        const int m = tb + j + half * 8;
                        if (m < rows) {
                            out[(size_t)f * N_T * 2 + (size_t)(start + m) * 2 + lrow] =
                                (po[j] + boutv) * lmf[j];
                        }
                    }
                }
            }
        }
        // No trailing barrier needed: gmbf/hbf are intra-wave; maskf/cxs/cys/red
        // hazards are covered by the loop-top and post-reduction barriers.
    }

    // -- last time slice is defined as zeros (d_out is poisoned) --
    for (int r = tid; r < rows; r += 256) {
        const size_t base = (size_t)(S_T - 1) * N_T * 2 + (size_t)(start + r) * 2;
        out[base + 0] = 0.0f;
        out[base + 1] = 0.0f;
    }
}

extern "C" void kernel_launch(void* const* d_in, const int* in_sizes, int n_in,
                              void* d_out, int out_size, void* d_ws, size_t ws_size,
                              hipStream_t stream) {
    (void)in_sizes; (void)n_in; (void)out_size; (void)d_ws; (void)ws_size;
    const float* nodes_abs = (const float*)d_in[0];
    // d_in[1] nodes_norm, d_in[2] shift_value, d_in[4] scenes: unused by reference math
    const float* seq_list  = (const float*)d_in[3];
    const int*   pednum    = (const int*)d_in[5];
    const float* W_in      = (const float*)d_in[6];
    const float* W_gm      = (const float*)d_in[7];
    const float* b_h       = (const float*)d_in[8];
    const float* W_out     = (const float*)d_in[9];
    const float* b_out     = (const float*)d_in[10];
    float* out = (float*)d_out;

    pfa_scan_kernel<<<dim3(B_T), dim3(256), SMEM_BYTES, stream>>>(
        nodes_abs, seq_list, pednum, W_in, W_gm, b_h, W_out, b_out, out);
}
